// GCN_64020782514379
// MI455X (gfx1250) — matrix operand order repaired and checked by reading, not compiled, not run on hardware
//
#include <hip/hip_runtime.h>
#include <stdint.h>

typedef __attribute__((ext_vector_type(16))) __bf16 v16bf;
typedef __attribute__((ext_vector_type(8)))  float  v8f;

#define N_NODES 20000
#define N_EDGES 160000
#define N_GRAPH 64

__device__ __forceinline__ unsigned short f2bf(float f) {
  unsigned int u = __float_as_uint(f);
  u += 0x7FFFu + ((u >> 16) & 1u);          // round-to-nearest-even
  return (unsigned short)(u >> 16);
}
__device__ __forceinline__ float bf2f(unsigned short h) {
  return __uint_as_float(((unsigned int)h) << 16);
}

union Frag { unsigned int u[8]; v16bf v; };

// ---------------- utility kernels ----------------
__global__ void k_zero_f32(float* __restrict__ p, int n) {
  int t = blockIdx.x * blockDim.x + threadIdx.x;
  if (t < n) p[t] = 0.f;
}

__global__ void k_cvt_x(const float* __restrict__ x, unsigned short* __restrict__ o, int n) {
  int t = blockIdx.x * blockDim.x + threadIdx.x;
  if (t < n) o[t] = f2bf(x[t]);
}

// W [din,dout] f32 -> Wt [dout,din] bf16 (transposed so B-fragment loads are packed pairs)
__global__ void k_wt_cvt(const float* __restrict__ W, unsigned short* __restrict__ Wt,
                         int din, int dout) {
  int t = blockIdx.x * blockDim.x + threadIdx.x;
  if (t >= din * dout) return;
  int o = t / din, i = t - o * din;
  Wt[t] = f2bf(W[(size_t)i * dout + o]);
}

__global__ void k_deg_count(const int* __restrict__ dst, float* __restrict__ deg) {
  int e = blockIdx.x * blockDim.x + threadIdx.x;
  if (e < N_EDGES) atomicAdd(&deg[dst[e]], 1.f);
}
__global__ void k_deg_inv(float* __restrict__ deg) {
  int n = blockIdx.x * blockDim.x + threadIdx.x;
  if (n < N_NODES) deg[n] = 1.f / fmaxf(deg[n], 1.f);
}

// agg[dst] += x[src]   (bf16 vector read, fp32 atomic accumulate; 4 features/thread)
__global__ void k_scatter(const unsigned short* __restrict__ xb,
                          const int* __restrict__ src, const int* __restrict__ dst,
                          float* __restrict__ agg, int din) {
  int cpe = din >> 2;
  int t = blockIdx.x * blockDim.x + threadIdx.x;
  if (t >= N_EDGES * cpe) return;
  int e = t / cpe, c = (t - e * cpe) << 2;
  int s = src[e], d = dst[e];
  uint2 q = *(const uint2*)(xb + (size_t)s * din + c);   // 4 packed bf16
  float* ap = agg + (size_t)d * din + c;
  atomicAdd(ap + 0, __uint_as_float(q.x << 16));
  atomicAdd(ap + 1, __uint_as_float(q.x & 0xFFFF0000u));
  atomicAdd(ap + 2, __uint_as_float(q.y << 16));
  atomicAdd(ap + 3, __uint_as_float(q.y & 0xFFFF0000u));
}

// aggB = bf16(aggF * invdeg[node])
__global__ void k_scale_cvt(const float* __restrict__ agg, const float* __restrict__ invd,
                            unsigned short* __restrict__ o, int din) {
  int t = blockIdx.x * blockDim.x + threadIdx.x;
  if (t >= N_NODES * din) return;
  o[t] = f2bf(agg[t] * invd[t / din]);
}

// ------ fused SAGE GEMM: out = act(A@Wl + X@Wr + b) ------
// Workgroup tile: 32 rows x 256 cols. 8 waves share one async-staged LDS A-chunk
// (32 rows x 32 K bf16 = 2 KB) per K-step; each wave owns a 32-col slice and
// streams its B fragments from L2. Per K-step per wave:
//   4 ds_load_b128 (A) + 4 global_load_b128 (B) -> 4 v_wmma_f32_16x16x32_bf16.
// A is staged via GLOBAL_LOAD_ASYNC_TO_LDS_B128 (ASYNCcnt) -> 8x A-reuse from LDS.
__global__ __launch_bounds__(256)
void k_sage_gemm(const unsigned short* __restrict__ A,   // agg  bf16 [N,din]
                 const unsigned short* __restrict__ X,   // x    bf16 [N,din]
                 const unsigned short* __restrict__ WlT, // bf16 [dout,din]
                 const unsigned short* __restrict__ WrT, // bf16 [dout,din]
                 const float* __restrict__ bias,         // f32  [dout]
                 unsigned short* __restrict__ out,       // bf16 [N,dout]
                 int din, int dout, int relu) {
  __shared__ __align__(16) unsigned int sA[32 * 16];     // 32 rows x 16 dwords (2 KB)

  const int tid  = threadIdx.x;
  const int lane = tid & 31;
  const int wave = tid >> 5;
  const int wgN  = dout >> 8;                            // column workgroups (1 or 2)
  const int tmWG = blockIdx.x / wgN;
  const int tnWG = blockIdx.x - tmWG * wgN;
  const int m0 = tmWG << 5;                              // 32-row block (exact: 625 blocks)
  const int n0 = (tnWG << 8) + (wave << 5);              // wave-private 32-col slice
  const int r = lane & 15, half = lane >> 4;

  // LDS byte offset of sA (shared aperture low 32 bits == LDS offset)
  const unsigned lds_base = (unsigned)(uintptr_t)(void*)sA;
  const int srow = tid >> 2, sseg = tid & 3;             // staging map: 128 threads
  const unsigned lds_dst = lds_base + (unsigned)(srow * 64 + sseg * 16);

  v8f acc00 = {}, acc01 = {}, acc10 = {}, acc11 = {};
  const int kd = din >> 1;                               // dwords (bf16 pairs) per row

#pragma unroll 1
  for (int pass = 0; pass < 2; ++pass) {
    const unsigned short* Ap = pass ? X : A;
    const unsigned short* Wp = pass ? WrT : WlT;
    const unsigned char* gA =
        (const unsigned char*)(Ap + (size_t)(m0 + srow) * din) + sseg * 16;
    const unsigned int* wb0 = (const unsigned int*)(Wp + (size_t)(n0 + r) * din);
    const unsigned int* wb1 = (const unsigned int*)(Wp + (size_t)(n0 + 16 + r) * din);
#pragma unroll 1
    for (int k0 = 0; k0 < kd; k0 += 16) {                // 32 K-elements per step
      if (tid < 128) {                                   // async stage A chunk -> LDS
        asm volatile("global_load_async_to_lds_b128 %0, %1, off"
                     :: "v"(lds_dst), "v"(gA + (size_t)k0 * 4)
                     : "memory");
      }
      asm volatile("s_wait_asynccnt 0" ::: "memory");
      __syncthreads();

      Frag a0, a1, b0, b1;
#pragma unroll
      for (int v = 0; v < 8; ++v) {
        const int o = (v & 3) + 4 * half + 8 * (v >> 2); // ISA A 16x32 fragment map
        a0.u[v] = sA[r * 16 + o];
        a1.u[v] = sA[(16 + r) * 16 + o];
        b0.u[v] = wb0[k0 + o];                           // ISA B 32x16 fragment map
        b1.u[v] = wb1[k0 + o];
      }
      acc00 = __builtin_amdgcn_wmma_f32_16x16x32_bf16(false, a0.v, false, b0.v, (short)0, acc00, false, false);
      acc01 = __builtin_amdgcn_wmma_f32_16x16x32_bf16(false, a0.v, false, b1.v, (short)0, acc01, false, false);
      acc10 = __builtin_amdgcn_wmma_f32_16x16x32_bf16(false, a1.v, false, b0.v, (short)0, acc10, false, false);
      acc11 = __builtin_amdgcn_wmma_f32_16x16x32_bf16(false, a1.v, false, b1.v, (short)0, acc11, false, false);

      __syncthreads();                                   // before next chunk overwrite
    }
  }

  const float bc0 = bias[n0 + r];
  const float bc1 = bias[n0 + 16 + r];
#pragma unroll
  for (int v = 0; v < 8; ++v) {                          // C layout: row v + 8*half, col = lane&15
    const size_t row0 = (size_t)(m0 + v + 8 * half) * dout;
    const size_t row1 = (size_t)(m0 + 16 + v + 8 * half) * dout;
    float y00 = acc00[v] + bc0, y01 = acc01[v] + bc1;
    float y10 = acc10[v] + bc0, y11 = acc11[v] + bc1;
    if (relu) {
      y00 = (y00 >= 0.f) ? y00 : 0.01f * y00;
      y01 = (y01 >= 0.f) ? y01 : 0.01f * y01;
      y10 = (y10 >= 0.f) ? y10 : 0.01f * y10;
      y11 = (y11 >= 0.f) ? y11 : 0.01f * y11;
    }
    out[row0 + n0 + r]      = f2bf(y00);
    out[row0 + n0 + 16 + r] = f2bf(y01);
    out[row1 + n0 + r]      = f2bf(y10);
    out[row1 + n0 + 16 + r] = f2bf(y11);
  }
}

// ---------------- pooling + head ----------------
__global__ void k_pool_count(const int* __restrict__ batch, float* __restrict__ cnt) {
  int n = blockIdx.x * blockDim.x + threadIdx.x;
  if (n < N_NODES) atomicAdd(&cnt[batch[n]], 1.f);
}
__global__ void k_pool_sum(const unsigned short* __restrict__ h,
                           const int* __restrict__ batch, float* __restrict__ pooled) {
  int t = blockIdx.x * blockDim.x + threadIdx.x;
  if (t >= N_NODES * 256) return;
  int n = t >> 8, f = t & 255;
  atomicAdd(&pooled[(size_t)batch[n] * 256 + f], bf2f(h[t]));
}
__global__ void k_head(const float* __restrict__ pooled, const float* __restrict__ cnt,
                       const float* __restrict__ Wlin, const float* __restrict__ blin,
                       float* __restrict__ out) {
  int id = threadIdx.x;
  if (id >= N_GRAPH * 6) return;
  int g = id / 6, j = id - g * 6;
  float inv = 1.f / fmaxf(cnt[g], 1.f);
  float s = 0.f;
  for (int f = 0; f < 256; ++f) s += pooled[g * 256 + f] * Wlin[f * 6 + j];
  out[id] = s * inv + blin[j];
}

// ---------------- host orchestration ----------------
extern "C" void kernel_launch(void* const* d_in, const int* in_sizes, int n_in,
                              void* d_out, int out_size, void* d_ws, size_t ws_size,
                              hipStream_t stream) {
  (void)in_sizes; (void)n_in; (void)out_size; (void)ws_size;
  const float* x     = (const float*)d_in[0];
  const int*   ei    = (const int*)d_in[1];
  const int*   src   = ei;
  const int*   dst   = ei + N_EDGES;
  const int*   batch = (const int*)d_in[2];
  static const int dims[6][2] = {{128,256},{256,256},{256,512},{512,512},{512,256},{256,256}};
  const float *Wl[6], *bl[6], *Wr[6];
  for (int l = 0; l < 6; ++l) {
    Wl[l] = (const float*)d_in[3 + 3 * l];
    bl[l] = (const float*)d_in[4 + 3 * l];
    Wr[l] = (const float*)d_in[5 + 3 * l];
  }
  const float* Wlin = (const float*)d_in[21];
  const float* blin = (const float*)d_in[22];
  float* out = (float*)d_out;

  // workspace carve-out
  char* ws = (char*)d_ws; size_t off = 0;
  auto carve = [&](size_t bytes) -> char* {
    char* p = ws + off; off = (off + bytes + 255) & ~(size_t)255; return p;
  };
  unsigned short* hA   = (unsigned short*)carve((size_t)N_NODES * 512 * 2);
  unsigned short* hB   = (unsigned short*)carve((size_t)N_NODES * 512 * 2);
  float*          aggF = (float*)         carve((size_t)N_NODES * 512 * 4);
  unsigned short* aggB = (unsigned short*)carve((size_t)N_NODES * 512 * 2);
  float*          deg  = (float*)         carve((size_t)N_NODES * 4);
  float*          pooled = (float*)       carve((size_t)N_GRAPH * 256 * 4);
  float*          cnt  = (float*)         carve((size_t)N_GRAPH * 4);
  unsigned short *WTl[6], *WTr[6];
  for (int l = 0; l < 6; ++l) {
    size_t ne = (size_t)dims[l][0] * dims[l][1];
    WTl[l] = (unsigned short*)carve(ne * 2);
    WTr[l] = (unsigned short*)carve(ne * 2);
  }

  const int TB = 256;
  auto blk = [](int n) { return dim3((unsigned)((n + 255) / 256)); };

  // convert input x -> bf16
  int nx = N_NODES * 128;
  k_cvt_x<<<blk(nx), TB, 0, stream>>>(x, hA, nx);

  // degrees -> inverse degrees
  k_zero_f32<<<blk(N_NODES), TB, 0, stream>>>(deg, N_NODES);
  k_deg_count<<<blk(N_EDGES), TB, 0, stream>>>(dst, deg);
  k_deg_inv<<<blk(N_NODES), TB, 0, stream>>>(deg);

  // pre-convert all weights to transposed bf16
  for (int l = 0; l < 6; ++l) {
    int ne = dims[l][0] * dims[l][1];
    k_wt_cvt<<<blk(ne), TB, 0, stream>>>(Wl[l], WTl[l], dims[l][0], dims[l][1]);
    k_wt_cvt<<<blk(ne), TB, 0, stream>>>(Wr[l], WTr[l], dims[l][0], dims[l][1]);
  }

  unsigned short* cur = hA;
  unsigned short* nxt = hB;
  for (int l = 0; l < 6; ++l) {
    const int din = dims[l][0], dout = dims[l][1];
    const int na = N_NODES * din;
    k_zero_f32<<<blk(na), TB, 0, stream>>>(aggF, na);
    k_scatter<<<blk(N_EDGES * (din >> 2)), TB, 0, stream>>>(cur, src, dst, aggF, din);
    k_scale_cvt<<<blk(na), TB, 0, stream>>>(aggF, deg, aggB, din);
    const int blocks = (N_NODES >> 5) * (dout >> 8);     // exact grid, no tail
    k_sage_gemm<<<dim3((unsigned)blocks), TB, 0, stream>>>(
        aggB, cur, WTl[l], WTr[l], bl[l], nxt, din, dout, (l < 5) ? 1 : 0);
    unsigned short* tmp = cur; cur = nxt; nxt = tmp;
  }

  // global mean pool + linear head
  k_zero_f32<<<blk(N_GRAPH * 256), TB, 0, stream>>>(pooled, N_GRAPH * 256);
  k_zero_f32<<<blk(N_GRAPH), TB, 0, stream>>>(cnt, N_GRAPH);
  k_pool_count<<<blk(N_NODES), TB, 0, stream>>>(batch, cnt);
  k_pool_sum<<<blk(N_NODES * 256), TB, 0, stream>>>(cur, batch, pooled);
  k_head<<<dim3(1), dim3(N_GRAPH * 6), 0, stream>>>(pooled, cnt, Wlin, blin, out);
}
